// DimeNetPlusPlus_P3M_4715874091791
// MI455X (gfx1250) — matrix operand order repaired and checked by reading, not compiled, not run on hardware
//
#include <hip/hip_runtime.h>
#include <hip/hip_bf16.h>

typedef __attribute__((ext_vector_type(16))) __bf16 v16bf;
typedef __attribute__((ext_vector_type(8)))  float  v8f;

#define H_DIM   128
#define INT_DIM 64
#define ROWS    32      // edge rows per block
#define ASTR    68      // act LDS row stride in dwords (136 bf16)
#define WSTR    130     // weight LDS pair-row stride in dwords
#define TPW     4       // triplets per wave in the scatter kernel

__device__ __forceinline__ float silu_f(float v) {
    return v * (1.0f / (1.0f + __expf(-v)));
}

// ---- LDS staging helpers -------------------------------------------------

// Global fp32 [rows x C] -> LDS bf16 row-major, row stride = strideBF halfwords
__device__ __forceinline__ void load_act_bf16(const float* __restrict__ X, int row0, int rows,
                                              int C, unsigned* s_act, int strideBF,
                                              int tid, int nthr, int Elim) {
    __bf16* h = (__bf16*)s_act;
    for (int idx = tid; idx < rows * C; idx += nthr) {
        int r = idx / C, c = idx - r * C;
        int rr = row0 + r; if (rr >= Elim) rr = Elim - 1;
        h[r * strideBF + c] = (__bf16)X[(size_t)rr * C + c];
    }
}

// Global fp32 weight [K x N] (row-major, N fastest) -> LDS packed bf16 pairs:
// dword at (k>>1)*WSTR + n holds (w[k][n], w[k+1][n]) in (lo,hi) halves.
__device__ __forceinline__ void load_weight_bf16(const float* __restrict__ W, int K, int N,
                                                 unsigned* s_w, int tid, int nthr) {
    __bf16* h = (__bf16*)s_w;
    for (int idx = tid; idx < K * N; idx += nthr) {
        int k = idx / N, n = idx - k * N;
        h[((k >> 1) * WSTR + n) * 2 + (k & 1)] = (__bf16)W[idx];
    }
}

// ---- WMMA fragment loaders (bf16 16x16x32) -------------------------------

// A: 16x32, lane holds row M=lane%16; VGPR0-3: K=half*8+{0..7}; VGPR4-7: K=16+half*8+{0..7}
__device__ __forceinline__ v16bf load_afrag(const unsigned* s_act, int strideDW,
                                            int rowbase, int k0, int lane) {
    int row = rowbase + (lane & 15);
    int half = lane >> 4;
    const unsigned* p = s_act + row * strideDW;
    union { v16bf v; unsigned u[8]; } f;
#pragma unroll
    for (int i = 0; i < 8; ++i) {
        int k = (i < 4) ? (k0 + half * 8 + i * 2) : (k0 + 16 + half * 8 + (i - 4) * 2);
        f.u[i] = p[k >> 1];
    }
    return f.v;
}

// B: 32x16, lane holds col N=lane%16; VGPR0-3: K=half*8+{0..7}; VGPR4-7: K=16+half*8+{0..7}
__device__ __forceinline__ v16bf load_bfrag(const unsigned* s_w, int n0, int k0, int lane) {
    int n = n0 + (lane & 15);
    int half = lane >> 4;
    union { v16bf v; unsigned u[8]; } f;
#pragma unroll
    for (int i = 0; i < 8; ++i) {
        int k = (i < 4) ? (k0 + half * 8 + i * 2) : (k0 + 16 + half * 8 + (i - 4) * 2);
        f.u[i] = s_w[(k >> 1) * WSTR + n];
    }
    return f.v;
}

// compute 2 adjacent 16x16 output tiles (same row-tile rt, cols ct0, ct0+1)
__device__ __forceinline__ void gemm2(const unsigned* act, const unsigned* w, int K,
                                      int rt, int ct0, int lane, float c[2][8]) {
#pragma unroll
    for (int j = 0; j < 2; ++j) {
        v8f acc = {};
        int n0 = (ct0 + j) * 16;
        for (int k0 = 0; k0 < K; k0 += 32) {
            v16bf a = load_afrag(act, ASTR, rt * 16, k0, lane);
            v16bf b = load_bfrag(w, n0, k0, lane);
            acc = __builtin_amdgcn_wmma_f32_16x16x32_bf16(false, a, false, b,
                                                          (short)0, acc, false, false);
        }
#pragma unroll
        for (int r = 0; r < 8; ++r) c[j][r] = acc[r];
    }
}

// ---- Kernel A: edge preprocessing ---------------------------------------
// x_ji = silu(x@W_ji+b); x_kj = silu(x@W_kj+b)*rbf_e; x_kj_down = silu(x_kj@W_down)
__global__ __launch_bounds__(256) void dime_edge_kernel(
    const float* __restrict__ x, const float* __restrict__ rbf,
    const float* __restrict__ W_rbf1, const float* __restrict__ W_rbf2,
    const float* __restrict__ W_kj, const float* __restrict__ b_kj,
    const float* __restrict__ W_ji, const float* __restrict__ b_ji,
    const float* __restrict__ W_down,
    float* __restrict__ xji_out, float* __restrict__ xkj_out, int E) {
    __shared__ unsigned s_w[64 * WSTR];
    __shared__ unsigned s_actA[ROWS * ASTR];
    __shared__ unsigned s_actB[ROWS * ASTR];
    __shared__ float    s_t1[ROWS * 8];
    __shared__ float    s_w2r[8 * 128];

    int tid = threadIdx.x;
    int lane = tid & 31, wave = tid >> 5;
    int row0 = blockIdx.x * ROWS;

    load_act_bf16(x, row0, ROWS, 128, s_actA, 136, tid, 256, E);
    {   // t1 = rbf @ W_rbf1  [32 x 8]
        int m = tid >> 3, b = tid & 7;
        int rr = row0 + m; if (rr >= E) rr = E - 1;
        float acc = 0.f;
#pragma unroll
        for (int r = 0; r < 6; ++r) acc += rbf[(size_t)rr * 6 + r] * W_rbf1[r * 8 + b];
        s_t1[m * 8 + b] = acc;
    }
    for (int i = tid; i < 8 * 128; i += 256) s_w2r[i] = W_rbf2[i];

    int rt = wave & 1, ct0 = (wave >> 1) * 2;
    int half = lane >> 4;

    // ---- stage 1: x_ji ----
    load_weight_bf16(W_ji, 128, 128, s_w, tid, 256);
    __syncthreads();
    {
        float c[2][8];
        gemm2(s_actA, s_w, 128, rt, ct0, lane, c);
#pragma unroll
        for (int j = 0; j < 2; ++j) {
            int col = (ct0 + j) * 16 + (lane & 15);
            float bias = b_ji[col];
#pragma unroll
            for (int r = 0; r < 8; ++r) {
                int m = rt * 16 + r + half * 8;
                if (row0 + m < E)
                    xji_out[(size_t)(row0 + m) * 128 + col] = silu_f(c[j][r] + bias);
            }
        }
    }
    __syncthreads();
    // ---- stage 2: x_kj * rbf_e -> LDS bf16 ----
    load_weight_bf16(W_kj, 128, 128, s_w, tid, 256);
    __syncthreads();
    {
        float c[2][8];
        gemm2(s_actA, s_w, 128, rt, ct0, lane, c);
        __bf16* hB = (__bf16*)s_actB;
#pragma unroll
        for (int j = 0; j < 2; ++j) {
            int col = (ct0 + j) * 16 + (lane & 15);
            float bias = b_kj[col];
#pragma unroll
            for (int r = 0; r < 8; ++r) {
                int m = rt * 16 + r + half * 8;
                float re = 0.f;
#pragma unroll
                for (int b = 0; b < 8; ++b) re += s_t1[m * 8 + b] * s_w2r[b * 128 + col];
                hB[m * 136 + col] = (__bf16)(silu_f(c[j][r] + bias) * re);
            }
        }
    }
    __syncthreads();
    // ---- stage 3: down projection 128 -> 64 ----
    load_weight_bf16(W_down, 128, 64, s_w, tid, 256);
    __syncthreads();
    {
        int ct = wave >> 1;          // 0..3 (64 output cols)
        int n0 = ct * 16;
        v8f acc = {};
        for (int k0 = 0; k0 < 128; k0 += 32) {
            v16bf a = load_afrag(s_actB, ASTR, rt * 16, k0, lane);
            v16bf b = load_bfrag(s_w, n0, k0, lane);
            acc = __builtin_amdgcn_wmma_f32_16x16x32_bf16(false, a, false, b,
                                                          (short)0, acc, false, false);
        }
        int col = n0 + (lane & 15);
#pragma unroll
        for (int r = 0; r < 8; ++r) {
            int m = rt * 16 + r + half * 8;
            if (row0 + m < E)
                xkj_out[(size_t)(row0 + m) * 64 + col] = silu_f(acc[r]);
        }
    }
}

// ---- Kernel B: triplet gather / sbf embed / scatter-add ------------------
// One wave per triplet, TPW triplets per wave; agg/xkj are L2-resident (128MB each).
__global__ __launch_bounds__(256) void dime_triplet_kernel(
    const float* __restrict__ sbf, const int* __restrict__ idx_kj,
    const int* __restrict__ idx_ji,
    const float* __restrict__ W_sbf1, const float* __restrict__ W_sbf2,
    const float* __restrict__ xkj, float* __restrict__ agg, int T) {
    __shared__ float s_w1[42 * 8];
    __shared__ float s_w2[8 * 64];
    int tid = threadIdx.x;
    for (int i = tid; i < 42 * 8; i += 256) s_w1[i] = W_sbf1[i];
    for (int i = tid; i < 8 * 64; i += 256) s_w2[i] = W_sbf2[i];
    __syncthreads();

    int wave = tid >> 5, lane = tid & 31;
    long long base = ((long long)blockIdx.x * 8 + wave) * TPW;

    // prefetch this wave's sbf rows (global_prefetch_b8)
    if (lane < TPW) {
        long long tp = base + lane;
        if (tp < T) __builtin_prefetch(sbf + tp * 42, 0, 3);
    }

#pragma unroll
    for (int it = 0; it < TPW; ++it) {
        long long t = base + it;
        if (t >= T) break;

        float t1 = 0.f;
        if (lane < 8) {
            const float* srow = sbf + t * 42;
#pragma unroll
            for (int r = 0; r < 42; ++r) t1 += srow[r] * s_w1[r * 8 + lane];
        }
        int kj = idx_kj[t], ji = idx_ji[t];
        int c0 = lane * 2;
        float e0 = 0.f, e1 = 0.f;
#pragma unroll
        for (int b = 0; b < 8; ++b) {
            float tb = __shfl(t1, b, 32);
            e0 += tb * s_w2[b * 64 + c0];
            e1 += tb * s_w2[b * 64 + c0 + 1];
        }
        const float2 xk = *(const float2*)(xkj + (size_t)kj * 64 + c0);
        float* dst = agg + (size_t)ji * 64 + c0;
        atomicAdd(dst,     xk.x * e0);
        atomicAdd(dst + 1, xk.y * e1);
    }
}

// ---- Kernel C helpers ----------------------------------------------------
__device__ __forceinline__ void residual_block(
    const float* __restrict__ W1, const float* __restrict__ b1,
    const float* __restrict__ W2, const float* __restrict__ b2,
    unsigned* s_w, unsigned* cur, unsigned* alt,
    float (&h)[2][8], int rt, int ct0, int lane, int tid) {
    int half = lane >> 4;
    load_weight_bf16(W1, 128, 128, s_w, tid, 256);
    __syncthreads();
    float c[2][8];
    gemm2(cur, s_w, 128, rt, ct0, lane, c);
    __bf16* ha = (__bf16*)alt;
#pragma unroll
    for (int j = 0; j < 2; ++j) {
        int col = (ct0 + j) * 16 + (lane & 15);
        float bias = b1[col];
#pragma unroll
        for (int r = 0; r < 8; ++r) {
            int m = rt * 16 + r + half * 8;
            ha[m * 136 + col] = (__bf16)silu_f(c[j][r] + bias);
        }
    }
    __syncthreads();
    load_weight_bf16(W2, 128, 128, s_w, tid, 256);
    __syncthreads();
    gemm2(alt, s_w, 128, rt, ct0, lane, c);
    __bf16* hc = (__bf16*)cur;
#pragma unroll
    for (int j = 0; j < 2; ++j) {
        int col = (ct0 + j) * 16 + (lane & 15);
        float bias = b2[col];
#pragma unroll
        for (int r = 0; r < 8; ++r) {
            int m = rt * 16 + r + half * 8;
            h[j][r] += silu_f(c[j][r] + bias);
            hc[m * 136 + col] = (__bf16)h[j][r];
        }
    }
    __syncthreads();
}

// ---- Kernel C: up-proj + residual stack ---------------------------------
__global__ __launch_bounds__(256) void dime_out_kernel(
    const float* __restrict__ x, const float* __restrict__ xji,
    const float* __restrict__ agg, const float* __restrict__ W_up,
    const float* __restrict__ Wb1, const float* __restrict__ bb1,
    const float* __restrict__ Wb2, const float* __restrict__ bb2,
    const float* __restrict__ W_lin, const float* __restrict__ b_lin,
    const float* __restrict__ Wa1, const float* __restrict__ ba1,
    const float* __restrict__ Wa2, const float* __restrict__ ba2,
    float* __restrict__ out, int E) {
    __shared__ unsigned s_w[64 * WSTR];
    __shared__ unsigned s_actA[ROWS * ASTR];
    __shared__ unsigned s_actB[ROWS * ASTR];

    int tid = threadIdx.x;
    int lane = tid & 31, wave = tid >> 5;
    int half = lane >> 4;
    int row0 = blockIdx.x * ROWS;
    int rt = wave & 1, ct0 = (wave >> 1) * 2;

    float h[2][8];

    // stage 0/1: agg -> bf16; h = xji + silu(agg @ W_up)
    load_act_bf16(agg, row0, ROWS, 64, s_actA, 136, tid, 256, E);
    load_weight_bf16(W_up, 64, 128, s_w, tid, 256);
    __syncthreads();
    {
        float c[2][8];
        gemm2(s_actA, s_w, 64, rt, ct0, lane, c);
        __bf16* hB = (__bf16*)s_actB;
#pragma unroll
        for (int j = 0; j < 2; ++j) {
            int col = (ct0 + j) * 16 + (lane & 15);
#pragma unroll
            for (int r = 0; r < 8; ++r) {
                int m = rt * 16 + r + half * 8;
                int rr = row0 + m; if (rr >= E) rr = E - 1;
                float v = xji[(size_t)rr * 128 + col] + silu_f(c[j][r]);
                h[j][r] = v;
                hB[m * 136 + col] = (__bf16)v;
            }
        }
    }
    __syncthreads();

    unsigned* cur = s_actB;
    unsigned* alt = s_actA;

    // num_before_skip = 1
    residual_block(Wb1, bb1, Wb2, bb2, s_w, cur, alt, h, rt, ct0, lane, tid);

    // h = silu(h @ W_lin + b_lin) + x
    load_weight_bf16(W_lin, 128, 128, s_w, tid, 256);
    __syncthreads();
    {
        float c[2][8];
        gemm2(cur, s_w, 128, rt, ct0, lane, c);
        __bf16* ha = (__bf16*)alt;
#pragma unroll
        for (int j = 0; j < 2; ++j) {
            int col = (ct0 + j) * 16 + (lane & 15);
            float bias = b_lin[col];
#pragma unroll
            for (int r = 0; r < 8; ++r) {
                int m = rt * 16 + r + half * 8;
                int rr = row0 + m; if (rr >= E) rr = E - 1;
                float v = silu_f(c[j][r] + bias) + x[(size_t)rr * 128 + col];
                h[j][r] = v;
                ha[m * 136 + col] = (__bf16)v;
            }
        }
    }
    __syncthreads();
    { unsigned* t = cur; cur = alt; alt = t; }

    // num_after_skip = 2
    residual_block(Wa1,              ba1,       Wa2,              ba2,       s_w, cur, alt, h, rt, ct0, lane, tid);
    residual_block(Wa1 + 128 * 128,  ba1 + 128, Wa2 + 128 * 128,  ba2 + 128, s_w, cur, alt, h, rt, ct0, lane, tid);

    // write final h
#pragma unroll
    for (int j = 0; j < 2; ++j) {
        int col = (ct0 + j) * 16 + (lane & 15);
#pragma unroll
        for (int r = 0; r < 8; ++r) {
            int m = rt * 16 + r + half * 8;
            if (row0 + m < E)
                out[(size_t)(row0 + m) * 128 + col] = h[j][r];
        }
    }
}

__global__ void dime_zero_kernel(float* __restrict__ p, long long n) {
    long long i = (long long)blockIdx.x * blockDim.x + threadIdx.x;
    if (i < n) p[i] = 0.f;
}

// ---- Host launch ---------------------------------------------------------
extern "C" void kernel_launch(void* const* d_in, const int* in_sizes, int n_in,
                              void* d_out, int out_size, void* d_ws, size_t ws_size,
                              hipStream_t stream) {
    const float* x      = (const float*)d_in[0];
    const float* rbf    = (const float*)d_in[1];
    const float* sbf    = (const float*)d_in[2];
    const int*   idx_kj = (const int*)d_in[3];
    const int*   idx_ji = (const int*)d_in[4];
    const float* W_rbf1 = (const float*)d_in[5];
    const float* W_rbf2 = (const float*)d_in[6];
    const float* W_sbf1 = (const float*)d_in[7];
    const float* W_sbf2 = (const float*)d_in[8];
    const float* W_kj   = (const float*)d_in[9];
    const float* b_kj   = (const float*)d_in[10];
    const float* W_ji   = (const float*)d_in[11];
    const float* b_ji   = (const float*)d_in[12];
    const float* W_down = (const float*)d_in[13];
    const float* W_up   = (const float*)d_in[14];
    const float* Wb1    = (const float*)d_in[15];
    const float* bb1    = (const float*)d_in[16];
    const float* Wb2    = (const float*)d_in[17];
    const float* bb2    = (const float*)d_in[18];
    const float* W_lin  = (const float*)d_in[19];
    const float* b_lin  = (const float*)d_in[20];
    const float* Wa1    = (const float*)d_in[21];
    const float* ba1    = (const float*)d_in[22];
    const float* Wa2    = (const float*)d_in[23];
    const float* ba2    = (const float*)d_in[24];

    const int E = in_sizes[1] / 6;   // rbf is [E, 6]
    const int T = in_sizes[3];       // idx_kj is [T]

    float* xji = (float*)d_ws;                      // [E,128]
    float* xkj = xji + (size_t)E * 128;             // [E,64]
    float* agg = xkj + (size_t)E * 64;              // [E,64]

    long long aggN = (long long)E * 64;
    dime_zero_kernel<<<(int)((aggN + 255) / 256), 256, 0, stream>>>(agg, aggN);

    int eblocks = (E + ROWS - 1) / ROWS;
    dime_edge_kernel<<<eblocks, 256, 0, stream>>>(
        x, rbf, W_rbf1, W_rbf2, W_kj, b_kj, W_ji, b_ji, W_down, xji, xkj, E);

    int tblocks = (int)(((long long)T + 8LL * TPW - 1) / (8LL * TPW));
    dime_triplet_kernel<<<tblocks, 256, 0, stream>>>(
        sbf, idx_kj, idx_ji, W_sbf1, W_sbf2, xkj, agg, T);

    dime_out_kernel<<<eblocks, 256, 0, stream>>>(
        x, xji, agg, W_up, Wb1, bb1, Wb2, bb2, W_lin, b_lin,
        Wa1, ba1, Wa2, ba2, (float*)d_out, E);
}
